// GraphSAGE_71760313581753
// MI455X (gfx1250) — matrix operand order repaired and checked by reading, not compiled, not run on hardware
//
#include <hip/hip_runtime.h>
#include <stdint.h>

typedef __attribute__((ext_vector_type(16))) __bf16   v16bf;
typedef __attribute__((ext_vector_type(8)))  float    v8f;
typedef __attribute__((ext_vector_type(4)))  uint32_t u32x4;
typedef __attribute__((ext_vector_type(8)))  uint32_t u32x8;

#define FD   128          // feature dim (D == H)
#define LDP  136          // padded LDS row stride (bf16 elems); 272B rows, 16B aligned

// ---------------------------------------------------------------------------
// CDNA5 async global->LDS copy (ASYNCcnt path, §15.18.3 op 98)
// ---------------------------------------------------------------------------
__device__ __forceinline__ uint32_t lds_off(const void* p) {
  // flat shared pointer: low 32 bits are the LDS address (ISA §10.2 aperture calc)
  return (uint32_t)(uintptr_t)p;
}
__device__ __forceinline__ void async_cp_b128(uint32_t lds, const void* gaddr) {
  asm volatile("global_load_async_to_lds_b128 %0, %1, off"
               :: "v"(lds), "v"(gaddr) : "memory");
}
__device__ __forceinline__ void async_wait0() {
  asm volatile("s_wait_asynccnt 0" ::: "memory");
}

// ---------------------------------------------------------------------------
// CDNA5 Tensor Data Mover: 2D tile (tile_dim0=128 bf16, tile_dim1=128 rows)
// from a (rows_remaining x 128) bf16 tensor into LDS with 8-bf16 row padding
// (pad_interval = 64 DWORDs -> one 256B row, pad_amount = 4 DWORDs -> LDP=136).
// OOB rows (beyond tensor_dim1) read as zero -> free ragged-tail handling.
// ---------------------------------------------------------------------------
__device__ __forceinline__ void tdm_load_tile_2d(uint32_t lds_base,
                                                 const void* gbase,
                                                 uint32_t rows_remaining) {
  uint64_t ga = (uint64_t)(uintptr_t)gbase;
  u32x4 g0;
  g0[0] = 1u;                                   // count=1 (valid user D#)
  g0[1] = lds_base;                             // lds_addr [63:32]
  g0[2] = (uint32_t)ga;                         // global_addr [95:64]
  g0[3] = (uint32_t)(ga >> 32) | (2u << 30);    // global_addr hi | type=2 (image)
  u32x8 g1;
  g1[0] = (1u << 16)                            // data_size = 2 bytes
        | (1u << 20)                            // pad_enable
        | (5u << 22)                            // pad_interval: 64 DWORDs
        | (3u << 25);                           // pad_amount : 4 DWORDs
  g1[1] = (128u & 0xFFFFu) << 16;               // tensor_dim0 lo16 (=128)
  g1[2] = (rows_remaining & 0xFFFFu) << 16;     // tensor_dim1 lo16 | dim0 hi16
  g1[3] = (128u << 16) | (rows_remaining >> 16);// tile_dim0=128 | tensor_dim1 hi16
  g1[4] = 128u;                                 // tile_dim1=128, tile_dim2=0
  g1[5] = 128u;                                 // tensor_dim0_stride lo32 = 128
  g1[6] = 0u;                                   // stride0 hi16 | stride1 lo16
  g1[7] = 0u;                                   // stride1 hi32
  asm volatile("tensor_load_to_lds %0, %1" :: "s"(g0), "s"(g1) : "memory");
}

// ---------------------------------------------------------------------------
// Utility kernels
// ---------------------------------------------------------------------------
__global__ void zero_f32_kernel(float* __restrict__ p, long long count) {
  long long i = (long long)blockIdx.x * blockDim.x + threadIdx.x;
  if (i < count) p[i] = 0.0f;
}

__global__ void f32_to_bf16_kernel(const float* __restrict__ s,
                                   __bf16* __restrict__ d, long long count) {
  long long i = (long long)blockIdx.x * blockDim.x + threadIdx.x;
  if (i < count) d[i] = (__bf16)s[i];
}

// mean-aggregation scatter: one thread handles 4 features of one edge
__global__ void edge_scatter_kernel(const float* __restrict__ feat,
                                    const long long* __restrict__ ei,
                                    long long E,
                                    float* __restrict__ aggr,
                                    float* __restrict__ deg) {
  long long t = (long long)blockIdx.x * blockDim.x + threadIdx.x;
  if (t >= E * 32) return;
  long long e = t >> 5;
  int c = (int)(t & 31) * 4;
  long long s  = ei[e];
  long long dn = ei[E + e];
  const float* xs = feat + s * FD + c;
  float* ad = aggr + dn * FD + c;
  atomicAdd(ad + 0, xs[0]);
  atomicAdd(ad + 1, xs[1]);
  atomicAdd(ad + 2, xs[2]);
  atomicAdd(ad + 3, xs[3]);
  if ((t & 31) == 0) atomicAdd(deg + dn, 1.0f);
}

// divide by degree, convert aggr and layer-input to bf16
__global__ void finalize_kernel(const float* __restrict__ aggr,
                                const float* __restrict__ deg,
                                const float* __restrict__ xin,
                                __bf16* __restrict__ abf,
                                __bf16* __restrict__ xbf, long long count) {
  long long i = (long long)blockIdx.x * blockDim.x + threadIdx.x;
  if (i >= count) return;
  int nrow = (int)(i >> 7);
  float d = fmaxf(deg[nrow], 1.0f);
  abf[i] = (__bf16)(aggr[i] / d);
  xbf[i] = (__bf16)xin[i];
}

// pre-pack weight (K x Nc, row-major f32) into WMMA bf16 B-fragment order:
// fragment(kt,nt): lane l, elem e holds W[kt*32 + (l>=16?8:0) + (e<8?e:e+8)][nt*16 + l%16]
__global__ void pack_w_kernel(const float* __restrict__ W,
                              __bf16* __restrict__ P, int K, int Nc) {
  int t = blockIdx.x * blockDim.x + threadIdx.x;
  int NT = Nc >> 4, KT = K >> 5;
  if (t >= KT * NT * 32) return;
  int lane = t & 31, tile = t >> 5;
  int kt = tile / NT, ntl = tile - kt * NT;
  int nn = ntl * 16 + (lane & 15);
  int kb = kt * 32 + ((lane >> 4) ? 8 : 0);
  __bf16* dst = P + (size_t)t * 16;
#pragma unroll
  for (int e = 0; e < 16; ++e)
    dst[e] = (__bf16)W[(size_t)(kb + (e < 8 ? e : e + 8)) * Nc + nn];
}

// ---------------------------------------------------------------------------
// Fused SAGE layer: h = aggr@Wl + bl + x@Wr ; LayerNorm ; ReLU ; out = x + h
// block = 256 threads (8 wave32), 128 rows per block, each wave: 16 rows x 128 cols
// Activations staged into LDS by the Tensor Data Mover (wave 0 issues 2 D#s).
// ---------------------------------------------------------------------------
__global__ __launch_bounds__(256)
void sage_gemm_ln_kernel(const float* __restrict__ xres,
                         const __bf16* __restrict__ Abf,
                         const __bf16* __restrict__ Xbf,
                         const __bf16* __restrict__ Wlp,
                         const __bf16* __restrict__ Wrp,
                         const float* __restrict__ bL,
                         const float* __restrict__ gam,
                         const float* __restrict__ bet,
                         float* __restrict__ out, int n) {
  __shared__ __bf16 sA[128 * LDP];
  __shared__ __bf16 sX[128 * LDP];
  const int tid  = threadIdx.x;
  const int lane = tid & 31;
  const int wave = tid >> 5;
  const int row0 = blockIdx.x * 128;

  // stage 128x128 bf16 tiles of aggr and x via the TDM (one descriptor each)
  if (wave == 0) {
    uint32_t ldsA = (uint32_t)__builtin_amdgcn_readfirstlane((int)lds_off(&sA[0]));
    uint32_t ldsX = (uint32_t)__builtin_amdgcn_readfirstlane((int)lds_off(&sX[0]));
    uint32_t rem  = (uint32_t)(n - row0);     // rows remaining; OOB rows read 0
    tdm_load_tile_2d(ldsA, Abf + (size_t)row0 * FD, rem);
    tdm_load_tile_2d(ldsX, Xbf + (size_t)row0 * FD, rem);
    __builtin_amdgcn_s_wait_tensorcnt(0);
  }
  __syncthreads();

  const int mrow   = wave * 16 + (lane & 15);       // A-fragment source row
  const int khalfA = (lane >> 4) ? 8 : 0;           // lanes 16-31: K offset +8

  v8f acc[8];
  v8f zz = {};
#pragma unroll
  for (int t = 0; t < 8; ++t) acc[t] = zz;

  // double-buffered packed-weight fragments: load (idx+1) while WMMA'ing idx
  v16bf wlc = *(const v16bf*)(Wlp + ((size_t)lane << 4));
  v16bf wrc = *(const v16bf*)(Wrp + ((size_t)lane << 4));
#pragma unroll
  for (int kt = 0; kt < 4; ++kt) {
    union { v16bf v; uint32_t u[8]; } fa, fx;
#pragma unroll
    for (int vg = 0; vg < 8; ++vg) {
      int kl = kt * 32 + ((vg & 4) ? 16 : 0) + khalfA + (vg & 3) * 2;
      fa.u[vg] = *(const uint32_t*)&sA[mrow * LDP + kl];
      fx.u[vg] = *(const uint32_t*)&sX[mrow * LDP + kl];
    }
#pragma unroll
    for (int nt = 0; nt < 8; ++nt) {
      const int idx = kt * 8 + nt;
      v16bf wln = wlc, wrn = wrc;
      if (idx < 31) {
        wln = *(const v16bf*)(Wlp + (((size_t)(idx + 1) * 32 + lane) << 4));
        wrn = *(const v16bf*)(Wrp + (((size_t)(idx + 1) * 32 + lane) << 4));
      }
      acc[nt] = __builtin_amdgcn_wmma_f32_16x16x32_bf16(
          false, fa.v, false, wlc, (short)0, acc[nt], false, false);
      acc[nt] = __builtin_amdgcn_wmma_f32_16x16x32_bf16(
          false, fx.v, false, wrc, (short)0, acc[nt], false, false);
      wlc = wln; wrc = wrn;
    }
  }

  // epilogue: bias, LayerNorm stats (reduce across each 16-lane half), ReLU, residual
  float s[8], q[8];
#pragma unroll
  for (int r = 0; r < 8; ++r) { s[r] = 0.0f; q[r] = 0.0f; }
#pragma unroll
  for (int nt = 0; nt < 8; ++nt) {
    int c = nt * 16 + (lane & 15);
    float bb = bL[c];
#pragma unroll
    for (int r = 0; r < 8; ++r) {
      float v = acc[nt][r] + bb;
      acc[nt][r] = v;
      s[r] += v;
      q[r] += v * v;
    }
  }
#pragma unroll
  for (int m = 1; m < 16; m <<= 1) {
#pragma unroll
    for (int r = 0; r < 8; ++r) {
      s[r] += __shfl_xor(s[r], m, 32);
      q[r] += __shfl_xor(q[r], m, 32);
    }
  }
  float mu[8], rs[8];
#pragma unroll
  for (int r = 0; r < 8; ++r) {
    mu[r] = s[r] * (1.0f / 128.0f);
    float var = q[r] * (1.0f / 128.0f) - mu[r] * mu[r];
    rs[r] = rsqrtf(var + 1e-5f);
  }
  const int rbase = row0 + wave * 16 + ((lane >> 4) ? 8 : 0);
#pragma unroll
  for (int nt = 0; nt < 8; ++nt) {
    int c = nt * 16 + (lane & 15);
    float gg = gam[c], tb = bet[c];
#pragma unroll
    for (int r = 0; r < 8; ++r) {
      int row = rbase + r;
      if (row < n) {
        float v = (acc[nt][r] - mu[r]) * rs[r] * gg + tb;
        v = fmaxf(v, 0.0f);
        out[(size_t)row * FD + c] = xres[(size_t)row * FD + c] + v;
      }
    }
  }
}

// ---------------------------------------------------------------------------
// Head: out = X @ W_head + b_head   (128 -> 64); async-to-LDS staging path
// ---------------------------------------------------------------------------
__global__ __launch_bounds__(256)
void head_gemm_kernel(const __bf16* __restrict__ Xbf,
                      const __bf16* __restrict__ Wp,
                      const float* __restrict__ bh,
                      float* __restrict__ out, int n) {
  __shared__ __bf16 sX[128 * LDP];
  const int tid  = threadIdx.x;
  const int lane = tid & 31;
  const int wave = tid >> 5;
  const int row0 = blockIdx.x * 128;

#pragma unroll
  for (int i = 0; i < 8; ++i) {
    int ch = tid + i * 256;
    int r  = ch >> 4;
    int co = (ch & 15) * 8;
    int gr = row0 + r; if (gr >= n) gr = n - 1;
    async_cp_b128(lds_off(&sX[r * LDP + co]), Xbf + (size_t)gr * FD + co);
  }
  async_wait0();
  __syncthreads();

  const int mrow   = wave * 16 + (lane & 15);
  const int khalfA = (lane >> 4) ? 8 : 0;

  v8f acc[4];
  v8f zz = {};
#pragma unroll
  for (int t = 0; t < 4; ++t) acc[t] = zz;

  v16bf wfc = *(const v16bf*)(Wp + ((size_t)lane << 4));
#pragma unroll
  for (int kt = 0; kt < 4; ++kt) {
    union { v16bf v; uint32_t u[8]; } fx;
#pragma unroll
    for (int vg = 0; vg < 8; ++vg) {
      int kl = kt * 32 + ((vg & 4) ? 16 : 0) + khalfA + (vg & 3) * 2;
      fx.u[vg] = *(const uint32_t*)&sX[mrow * LDP + kl];
    }
#pragma unroll
    for (int nt = 0; nt < 4; ++nt) {
      const int idx = kt * 4 + nt;
      v16bf wfn = wfc;
      if (idx < 15)
        wfn = *(const v16bf*)(Wp + (((size_t)(idx + 1) * 32 + lane) << 4));
      acc[nt] = __builtin_amdgcn_wmma_f32_16x16x32_bf16(
          false, fx.v, false, wfc, (short)0, acc[nt], false, false);
      wfc = wfn;
    }
  }

  const int rbase = row0 + wave * 16 + ((lane >> 4) ? 8 : 0);
#pragma unroll
  for (int nt = 0; nt < 4; ++nt) {
    int c = nt * 16 + (lane & 15);
    float bb = bh[c];
#pragma unroll
    for (int r = 0; r < 8; ++r) {
      int row = rbase + r;
      if (row < n) out[(size_t)row * 64 + c] = acc[nt][r] + bb;
    }
  }
}

// ---------------------------------------------------------------------------
// Host launcher
// ---------------------------------------------------------------------------
extern "C" void kernel_launch(void* const* d_in, const int* in_sizes, int n_in,
                              void* d_out, int out_size, void* d_ws, size_t ws_size,
                              hipStream_t stream) {
  const float* x       = (const float*)d_in[0];
  const long long* ei  = (const long long*)d_in[1];
  const float* W_l0    = (const float*)d_in[2];
  const float* b_l0    = (const float*)d_in[3];
  const float* W_r0    = (const float*)d_in[4];
  const float* W_l1    = (const float*)d_in[5];
  const float* b_l1    = (const float*)d_in[6];
  const float* W_r1    = (const float*)d_in[7];
  const float* g0      = (const float*)d_in[8];
  const float* bt0     = (const float*)d_in[9];
  const float* g1      = (const float*)d_in[10];
  const float* bt1     = (const float*)d_in[11];
  const float* W_head  = (const float*)d_in[12];
  const float* b_head  = (const float*)d_in[13];
  float* out = (float*)d_out;

  const long long N  = in_sizes[0] / FD;
  const long long E  = in_sizes[1] / 2;
  const long long NF = N * FD;

  // carve workspace
  char* w = (char*)d_ws;
  auto carve = [&](size_t bytes) -> char* {
    char* p = w; w += (bytes + 255) & ~(size_t)255; return p;
  };
  float*  aggr  = (float*) carve((size_t)NF * 4);
  float*  deg   = (float*) carve((size_t)N * 4);
  __bf16* abf   = (__bf16*)carve((size_t)NF * 2);
  __bf16* xbf   = (__bf16*)carve((size_t)NF * 2);
  float*  h1    = (float*) carve((size_t)NF * 4);
  float*  h2    = (float*) carve((size_t)NF * 4);
  __bf16* h2bf  = (__bf16*)carve((size_t)NF * 2);
  __bf16* Wl0p  = (__bf16*)carve(128 * 128 * 2);
  __bf16* Wr0p  = (__bf16*)carve(128 * 128 * 2);
  __bf16* Wl1p  = (__bf16*)carve(128 * 128 * 2);
  __bf16* Wr1p  = (__bf16*)carve(128 * 128 * 2);
  __bf16* Whp   = (__bf16*)carve(128 * 64 * 2);

  const int TB = 256;
  // pack weights into WMMA B-fragment layout
  pack_w_kernel<<<(4 * 8 * 32 + TB - 1) / TB, TB, 0, stream>>>(W_l0, Wl0p, 128, 128);
  pack_w_kernel<<<(4 * 8 * 32 + TB - 1) / TB, TB, 0, stream>>>(W_r0, Wr0p, 128, 128);
  pack_w_kernel<<<(4 * 8 * 32 + TB - 1) / TB, TB, 0, stream>>>(W_l1, Wl1p, 128, 128);
  pack_w_kernel<<<(4 * 8 * 32 + TB - 1) / TB, TB, 0, stream>>>(W_r1, Wr1p, 128, 128);
  pack_w_kernel<<<(4 * 4 * 32 + TB - 1) / TB, TB, 0, stream>>>(W_head, Whp, 128, 64);

  const unsigned gNF   = (unsigned)((NF + TB - 1) / TB);
  const unsigned gN    = (unsigned)((N + TB - 1) / TB);
  const unsigned gEdge = (unsigned)((E * 32 + TB - 1) / TB);
  const unsigned gRows = (unsigned)((N + 127) / 128);

  // ----- layer 0 -----
  zero_f32_kernel<<<gNF, TB, 0, stream>>>(aggr, NF);
  zero_f32_kernel<<<gN,  TB, 0, stream>>>(deg, N);
  edge_scatter_kernel<<<gEdge, TB, 0, stream>>>(x, ei, E, aggr, deg);
  finalize_kernel<<<gNF, TB, 0, stream>>>(aggr, deg, x, abf, xbf, NF);
  sage_gemm_ln_kernel<<<gRows, TB, 0, stream>>>(x, abf, xbf, Wl0p, Wr0p,
                                                b_l0, g0, bt0, h1, (int)N);
  // ----- layer 1 -----
  zero_f32_kernel<<<gNF, TB, 0, stream>>>(aggr, NF);
  zero_f32_kernel<<<gN,  TB, 0, stream>>>(deg, N);
  edge_scatter_kernel<<<gEdge, TB, 0, stream>>>(h1, ei, E, aggr, deg);
  finalize_kernel<<<gNF, TB, 0, stream>>>(aggr, deg, h1, abf, xbf, NF);
  sage_gemm_ln_kernel<<<gRows, TB, 0, stream>>>(h1, abf, xbf, Wl1p, Wr1p,
                                                b_l1, g1, bt1, h2, (int)N);
  // ----- head -----
  f32_to_bf16_kernel<<<gNF, TB, 0, stream>>>(h2, h2bf, NF);
  head_gemm_kernel<<<gRows, TB, 0, stream>>>(h2bf, Whp, b_head, out, (int)N);
}
